// perceiver_resampler_15083925143914
// MI455X (gfx1250) — compile-verified, matrix operand order checked by reading, not run on hardware
//
#include <hip/hip_runtime.h>

// ---------------------------------------------------------------------------
// Perceiver-resampler cross attention for MI455X (gfx1250, wave32, WMMA).
//   kernel 1: K = x @ Wk^T + bk  -> bf16, layout [b*T+t][1024]  (d-contig)
//             V = x @ Wv^T + bv  -> bf16, layout [b*1024+n][T]  (t-contig)
//             (double-buffered LDS, software-pipelined global loads)
//   kernel 2: per (b, 16-l tile) block, 16 waves = 16 heads sharing the
//             ch_mask tile in LDS; online-softmax flash attention with
//             v_wmma_f32_16x16x32_bf16 for both S = lq*K^T and O += P*V.
// ---------------------------------------------------------------------------

#define B_   4
#define T_   1024
#define D_   1024
#define H_   16
#define DH_  64
#define L_   1344   // 21 * 64

typedef __attribute__((ext_vector_type(16))) __bf16 v16bf;
typedef __attribute__((ext_vector_type(8)))  __bf16 v8bf;
typedef __attribute__((ext_vector_type(4)))  __bf16 v4bf;
typedef __attribute__((ext_vector_type(8)))  float  v8f;

union Frag16 { v16bf v; v8bf h[2]; };

static __device__ __forceinline__ v8f wmma_bf16(const Frag16& A, const Frag16& Bm, v8f C) {
  // 8 args: (neg_a, A, neg_b, B, c_mod, C, reuse_a, reuse_b)
  return __builtin_amdgcn_wmma_f32_16x16x32_bf16(false, A.v, false, Bm.v, (short)0, C, false, false);
}

static __device__ __forceinline__ v8bf load8_f32_bf16(const float* __restrict__ s, float scale) {
  float4 a = *(const float4*)(s);
  float4 b = *(const float4*)(s + 4);
  v8bf r;
  r[0] = (__bf16)(a.x * scale); r[1] = (__bf16)(a.y * scale);
  r[2] = (__bf16)(a.z * scale); r[3] = (__bf16)(a.w * scale);
  r[4] = (__bf16)(b.x * scale); r[5] = (__bf16)(b.y * scale);
  r[6] = (__bf16)(b.z * scale); r[7] = (__bf16)(b.w * scale);
  return r;
}

static __device__ __forceinline__ v4bf cvt4(float4 u) {
  v4bf p;
  p[0] = (__bf16)u.x; p[1] = (__bf16)u.y; p[2] = (__bf16)u.z; p[3] = (__bf16)u.w;
  return p;
}

// ---------------------------------------------------------------------------
// Kernel 1: K/V projection GEMM.  grid = (BT/64, D/64, 2), block = 128 (4 waves)
// Each wave owns a 16(row) x 64(col) strip -> 4 fp32 16x16 accumulators.
// Double-buffered LDS (stride 40 halves = 20 banks, conflict-free, 16B-aligned
// fragment reads); next chunk's global loads issued before the barrier so they
// overlap the WMMA block; one barrier per k-step.
// ---------------------------------------------------------------------------
__global__ __launch_bounds__(128) void proj_kv_kernel(
    const float* __restrict__ x,
    const float* __restrict__ wk, const float* __restrict__ bk,
    const float* __restrict__ wv, const float* __restrict__ bv,
    __bf16* __restrict__ Kws, __bf16* __restrict__ Vt)
{
  __shared__ __bf16 ldsA[2][64 * 40];
  __shared__ __bf16 ldsB[2][64 * 40];

  const int tid   = threadIdx.x;
  const int lane  = tid & 31;
  const int wave  = tid >> 5;
  const int bt0   = blockIdx.x * 64;
  const int n0    = blockIdx.y * 64;
  const bool isK  = (blockIdx.z == 0);
  const float* W    = isK ? wk : wv;
  const float* bias = isK ? bk : bv;

  const int srow  = tid >> 1;          // 0..63
  const int shalf = (tid & 1) * 16;    // 0 or 16
  const float* xsrc = x + (size_t)(bt0 + srow) * D_ + shalf;
  const float* wsrc = W + (size_t)(n0 + srow) * D_ + shalf;

  v8f acc[4];
#pragma unroll
  for (int nt = 0; nt < 4; ++nt) acc[nt] = (v8f)0.0f;

  float4 rx[4], rw[4];
  // ---- prologue: load + stage chunk 0 ----
#pragma unroll
  for (int i = 0; i < 4; ++i) {
    rx[i] = ((const float4*)xsrc)[i];
    rw[i] = ((const float4*)wsrc)[i];
  }
  {
    __bf16* dA = &ldsA[0][srow * 40 + shalf];
    __bf16* dB = &ldsB[0][srow * 40 + shalf];
#pragma unroll
    for (int i = 0; i < 4; ++i) {
      ((v4bf*)dA)[i] = cvt4(rx[i]);
      ((v4bf*)dB)[i] = cvt4(rw[i]);
    }
  }

  const int frow = lane & 15;
  const int run0 = (lane < 16) ? 0 : 8;
  int cur = 0;

  for (int kc = 0; kc < D_; kc += 32) {
    const bool has_next = (kc + 32) < D_;
    // issue next chunk's global loads early (overlap with WMMA below)
    if (has_next) {
#pragma unroll
      for (int i = 0; i < 4; ++i) {
        rx[i] = ((const float4*)(xsrc + kc + 32))[i];
        rw[i] = ((const float4*)(wsrc + kc + 32))[i];
      }
    }
    __syncthreads();   // buf[cur] staged by all waves

    // ---- fragments + WMMA from buf[cur] ----
    Frag16 a;
    a.h[0] = *(const v8bf*)(&ldsA[cur][(wave * 16 + frow) * 40 + run0]);
    a.h[1] = *(const v8bf*)(&ldsA[cur][(wave * 16 + frow) * 40 + run0 + 16]);
#pragma unroll
    for (int nt = 0; nt < 4; ++nt) {
      Frag16 bfrag;
      bfrag.h[0] = *(const v8bf*)(&ldsB[cur][(nt * 16 + frow) * 40 + run0]);
      bfrag.h[1] = *(const v8bf*)(&ldsB[cur][(nt * 16 + frow) * 40 + run0 + 16]);
      acc[nt] = wmma_bf16(a, bfrag, acc[nt]);
    }

    // stage next chunk into the alternate buffer (no second barrier needed)
    if (has_next) {
      __bf16* dA = &ldsA[cur ^ 1][srow * 40 + shalf];
      __bf16* dB = &ldsB[cur ^ 1][srow * 40 + shalf];
#pragma unroll
      for (int i = 0; i < 4; ++i) {
        ((v4bf*)dA)[i] = cvt4(rx[i]);
        ((v4bf*)dB)[i] = cvt4(rw[i]);
      }
    }
    cur ^= 1;
  }

  // ---- epilogue: +bias, store bf16 (uniform branch hoisted) ----
  const int ncol  = lane & 15;
  const int mbase = (lane < 16) ? 0 : 8;
  if (isK) {
#pragma unroll
    for (int nt = 0; nt < 4; ++nt) {
      const int n = n0 + nt * 16 + ncol;
      const float bval = bias[n];
#pragma unroll
      for (int r = 0; r < 8; ++r) {
        const int m = r + mbase;
        Kws[(size_t)(bt0 + wave * 16 + m) * D_ + n] = (__bf16)(acc[nt][r] + bval);
      }
    }
  } else {
    const int b     = bt0 / T_;
    const int tbase = (bt0 % T_) + wave * 16;
#pragma unroll
    for (int nt = 0; nt < 4; ++nt) {
      const int n = n0 + nt * 16 + ncol;
      const float bval = bias[n];
#pragma unroll
      for (int r = 0; r < 8; ++r) {
        const int m = r + mbase;
        // transposed V: row = b*1024 + n, col = t  (t-contiguous rows)
        Vt[((size_t)(b * D_ + n)) * T_ + (tbase + m)] = (__bf16)(acc[nt][r] + bval);
      }
    }
  }
}

// ---------------------------------------------------------------------------
// Kernel 2: flash attention.  grid = (L/16, B), block = 512 = 16 waves.
// wave w == head h; ch_mask tile (16 x 1024 bytes) shared by all heads in LDS.
// Per 32-wide t chunk: batch-load 4 K fragments -> 4 WMMAs for S, online
// softmax with 16-lane shfl reductions, P staged per-wave through LDS into
// A-layout, batch-load 4 V fragments -> 4 WMMAs for O.
// ---------------------------------------------------------------------------
__global__ __launch_bounds__(512) void attn_kernel(
    const unsigned char* __restrict__ mask,
    const float* __restrict__ lq,
    const __bf16* __restrict__ Kws,
    const __bf16* __restrict__ Vt,
    float* __restrict__ out)
{
  __shared__ alignas(16) unsigned char smask[16][1024];
  __shared__ __bf16 sP[16][16 * 40];   // per-wave 16x32 P tile (stride 40)

  const int tid  = threadIdx.x;
  const int lane = tid & 31;
  const int h    = tid >> 5;           // head == wave
  const int l0   = blockIdx.x * 16;
  const int b    = blockIdx.y;

  // ---- cooperative mask tile load (head-independent, shared) ----
  {
    const int row = tid >> 5;
    const int seg = tid & 31;
    const int4* src = (const int4*)(mask + ((size_t)(b * L_ + l0 + row)) * T_ + seg * 32);
    int4* dst = (int4*)(&smask[row][seg * 32]);
    dst[0] = src[0];
    dst[1] = src[1];
  }
  __syncthreads();

  const int nlo   = lane & 15;
  const int run0  = (lane < 16) ? 0 : 8;
  const int mbase = (lane < 16) ? 0 : 8;

  // ---- lq A-fragments, softmax scale H^-0.5 = 0.25 folded in (exact in bf16) ----
  Frag16 lqf[2];
#pragma unroll
  for (int f = 0; f < 2; ++f) {
    const float* base = lq + (size_t)h * (L_ * DH_) + (size_t)(l0 + nlo) * DH_ + f * 32 + run0;
    lqf[f].h[0] = load8_f32_bf16(base,      0.25f);
    lqf[f].h[1] = load8_f32_bf16(base + 16, 0.25f);
  }

  float m_run[8], l_run[8];
  v8f O[4];
#pragma unroll
  for (int r = 0; r < 8; ++r) { m_run[r] = -3.0e38f; l_run[r] = 0.0f; }
#pragma unroll
  for (int dt = 0; dt < 4; ++dt) O[dt] = (v8f)0.0f;

  const __bf16* Kb = Kws + (size_t)(b * T_) * D_;
  const __bf16* Vb = Vt + (size_t)(b * D_ + h * DH_) * T_;

  for (int t0 = 0; t0 < T_; t0 += 32) {
    // ---- batch-load all 4 K fragments (loads issue back-to-back) ----
    Frag16 kf[2][2];
#pragma unroll
    for (int half = 0; half < 2; ++half) {
      const int t = t0 + half * 16 + nlo;
      const __bf16* kp = Kb + (size_t)t * D_ + h * DH_ + run0;
#pragma unroll
      for (int f = 0; f < 2; ++f) {
        kf[half][f].h[0] = *(const v8bf*)(kp + f * 32);
        kf[half][f].h[1] = *(const v8bf*)(kp + f * 32 + 16);
      }
    }
    // ---- S = (0.25*lq) @ K^T over d = 0..63 ----
    v8f S0 = (v8f)0.0f, S1 = (v8f)0.0f;
    S0 = wmma_bf16(lqf[0], kf[0][0], S0);
    S0 = wmma_bf16(lqf[1], kf[0][1], S0);
    S1 = wmma_bf16(lqf[0], kf[1][0], S1);
    S1 = wmma_bf16(lqf[1], kf[1][1], S1);

    // ---- masked online softmax (per C-layout row r) ----
#pragma unroll
    for (int r = 0; r < 8; ++r) {
      const int m = r + mbase;
      const bool mk0 = smask[m][t0 + nlo]      != 0;
      const bool mk1 = smask[m][t0 + 16 + nlo] != 0;
      float s0 = mk0 ? S0[r] : -1.0e9f;
      float s1 = mk1 ? S1[r] : -1.0e9f;
      float rm = fmaxf(s0, s1);
#pragma unroll
      for (int off = 8; off >= 1; off >>= 1) rm = fmaxf(rm, __shfl_xor(rm, off, 16));
      const float mn = fmaxf(m_run[r], rm);
      float e0 = mk0 ? __expf(s0 - mn) : 0.0f;   // re-zero masked lanes (matches ref)
      float e1 = mk1 ? __expf(s1 - mn) : 0.0f;
      float rs = e0 + e1;
#pragma unroll
      for (int off = 8; off >= 1; off >>= 1) rs += __shfl_xor(rs, off, 16);
      const float alpha = __expf(m_run[r] - mn);
      l_run[r] = l_run[r] * alpha + rs;
      m_run[r] = mn;
#pragma unroll
      for (int dt = 0; dt < 4; ++dt) O[dt][r] *= alpha;
      sP[h][m * 40 + nlo]      = (__bf16)e0;
      sP[h][m * 40 + 16 + nlo] = (__bf16)e1;
    }

    // ---- batch-load all 4 V fragments (overlap with LDS P reads) ----
    Frag16 vf[4];
#pragma unroll
    for (int dt = 0; dt < 4; ++dt) {
      const __bf16* vp = Vb + (size_t)(dt * 16 + nlo) * T_ + t0 + run0;
      vf[dt].h[0] = *(const v8bf*)(vp);
      vf[dt].h[1] = *(const v8bf*)(vp + 16);
    }

    // ---- P tile back out of LDS in A-layout (wave-local, LDS is in-order) ----
    Frag16 pf;
    pf.h[0] = *(const v8bf*)(&sP[h][nlo * 40 + run0]);
    pf.h[1] = *(const v8bf*)(&sP[h][nlo * 40 + run0 + 16]);

    // ---- O += P @ V  (V transposed: lane n = d, runs over t contiguous) ----
#pragma unroll
    for (int dt = 0; dt < 4; ++dt) O[dt] = wmma_bf16(pf, vf[dt], O[dt]);

    // prefetch next K chunk for this lane's t column
    if (t0 + 32 < T_) {
      __builtin_prefetch(Kb + (size_t)(t0 + 32 + nlo) * D_ + h * DH_, 0, 1);
    }
  }

  // ---- normalize + write (fully-masked rows -> l_run==0 -> 0, matches ref) ----
#pragma unroll
  for (int r = 0; r < 8; ++r) {
    const int m = r + mbase;
    const float inv = (l_run[r] > 0.0f) ? (1.0f / l_run[r]) : 0.0f;
#pragma unroll
    for (int dt = 0; dt < 4; ++dt) {
      out[((size_t)(b * L_ + l0 + m)) * D_ + h * DH_ + dt * 16 + nlo] = O[dt][r] * inv;
    }
  }
}

// ---------------------------------------------------------------------------
extern "C" void kernel_launch(void* const* d_in, const int* in_sizes, int n_in,
                              void* d_out, int out_size, void* d_ws, size_t ws_size,
                              hipStream_t stream) {
  (void)in_sizes; (void)n_in; (void)out_size; (void)ws_size;
  const float*         x    = (const float*)d_in[0];
  const unsigned char* msk  = (const unsigned char*)d_in[1];   // jax bool -> 1 byte
  const float*         lq   = (const float*)d_in[2];
  const float*         wk   = (const float*)d_in[3];
  const float*         bk   = (const float*)d_in[4];
  const float*         wv   = (const float*)d_in[5];
  const float*         bv   = (const float*)d_in[6];
  float*               out  = (float*)d_out;

  __bf16* Kws = (__bf16*)d_ws;                                  // B*T*D bf16 = 8 MB
  __bf16* Vt  = Kws + (size_t)B_ * T_ * D_;                     // B*D*T bf16 = 8 MB

  dim3 g1(B_ * T_ / 64, D_ / 64, 2), b1(128);
  proj_kv_kernel<<<g1, b1, 0, stream>>>(x, wk, bk, wv, bv, Kws, Vt);

  dim3 g2(L_ / 16, B_), b2(512);
  attn_kernel<<<g2, b2, 0, stream>>>(msk, lq, Kws, Vt, out);
}
